// L2Q_61289183314192
// MI455X (gfx1250) — matrix operand order repaired and checked by reading, not compiled, not run on hardware
//
#include <hip/hip_runtime.h>
#include <hip/hip_bf16.h>

// L2Q polynomial attention, flash-style single pass, bf16 WMMA on gfx1250.
// B=8 H=12 N=1024 D=64. Block = 128 threads (4 wave32), 64 query rows/block.
// Double-buffered K/V tiles in LDS + global_prefetch of tile t+2.

#define B_ 8
#define H_ 12
#define N_ 1024
#define D_ 64
#define SCALE_ 0.125f
#define EPS_ 1e-6f

typedef __bf16 bf16;
typedef __attribute__((ext_vector_type(16))) __bf16 bf16x16;
typedef __attribute__((ext_vector_type(8)))  __bf16 bf16x8;
typedef __attribute__((ext_vector_type(4)))  __bf16 bf16x4;
typedef __attribute__((ext_vector_type(8)))  float  f32x8;
typedef __attribute__((ext_vector_type(4)))  float  f32x4;

static __device__ inline bf16x16 cat16(bf16x8 lo, bf16x8 hi) {
    union { bf16x8 h[2]; bf16x16 v; } u;
    u.h[0] = lo; u.h[1] = hi;
    return u.v;
}
static __device__ inline bf16x8 cvt8(const float* p) {
    bf16x8 r;
#pragma unroll
    for (int i = 0; i < 8; ++i) r[i] = (__bf16)p[i];
    return r;
}
static __device__ inline bf16x8 lds8(const bf16* p) { return *(const bf16x8*)p; }

__global__ __launch_bounds__(128)
void l2q_attn_kernel(const float* __restrict__ q, const float* __restrict__ k,
                     const float* __restrict__ v, const float* __restrict__ alpha,
                     const float* __restrict__ beta, const float* __restrict__ gamma,
                     float* __restrict__ out)
{
    // LDS: 2x K tile bf16 [n][d] (2*8KB), 2x V tile bf16 transposed [d][n]
    // (2*8KB), per-wave P staging 16x64 bf16 (4*2KB). 40KB of the WGP's 320KB.
    __shared__ bf16 sK [2][64 * 64];
    __shared__ bf16 sVt[2][64 * 64];
    __shared__ bf16 sP [4][16 * 64];

    const int tid  = threadIdx.x;
    const int w    = tid >> 5;        // wave id 0..3
    const int lane = tid & 31;
    const int l15  = lane & 15;
    const int hi   = lane >> 4;       // 0: lanes 0-15, 1: lanes 16-31

    const int mt = blockIdx.x;        // query row-tile (0..15)
    const int h  = blockIdx.y;
    const int b  = blockIdx.z;

    const long bh = (long)b * H_ + h;
    const float* qb = q   + bh * (long)N_ * D_;
    const float* kb = k   + bh * (long)N_ * D_;
    const float* vb = v   + bh * (long)N_ * D_;
    float*       ob = out + bh * (long)N_ * D_;

    const float al = alpha[h], be = beta[h], ga = gamma[h];

    // ---- Q A-fragments (16-bit A 16x32 layout), loaded once, f32 -> bf16 ----
    const int   mRow = mt * 64 + w * 16 + l15;         // this lane's query row
    const float* qrow = qb + (long)mRow * D_;
    const int   c0 = hi * 8;                           // K sub-offset per half-wave
    bf16x16 aQ[2];
#pragma unroll
    for (int dk = 0; dk < 2; ++dk)                      // D=64 -> two K=32 chunks
        aQ[dk] = cat16(cvt8(qrow + dk * 32 + c0),
                       cvt8(qrow + dk * 32 + 16 + c0));

    f32x8 O[4];                                         // 16x64 output strip
#pragma unroll
    for (int i = 0; i < 4; ++i) O[i] = (f32x8){0,0,0,0,0,0,0,0};
    f32x8 rs = (f32x8){0,0,0,0,0,0,0,0};                // per-lane partial row sums

    // Cooperative stage of one 64x64 K/V tile into LDS buffer `buf`
    // (f32 -> bf16; V stored transposed for contiguous B-fragment reads).
    auto stage = [&](int buf, int nb) {
#pragma unroll
        for (int i = 0; i < 8; ++i) {
            const int e = (i * 128 + tid) * 4;          // element 0..4095, step 4
            const int n = e >> 6, d = e & 63;
            f32x4 kf = *(const f32x4*)(kb + (long)(nb + n) * D_ + d);
            bf16x4 k4;
#pragma unroll
            for (int j = 0; j < 4; ++j) k4[j] = (__bf16)kf[j];
            *(bf16x4*)&sK[buf][n * 64 + d] = k4;

            f32x4 vf = *(const f32x4*)(vb + (long)(nb + n) * D_ + d);
#pragma unroll
            for (int j = 0; j < 4; ++j) sVt[buf][(d + j) * 64 + n] = (__bf16)vf[j];
        }
    };

    stage(0, 0);                                        // prologue: tile 0

    constexpr int NT = N_ / 64;
    for (int it = 0; it < NT; ++it) {
        const int nb = it * 64;
        __syncthreads();   // buf[it&1] staged; prior readers of buf[(it+1)&1] done

        if (it + 2 < NT) {
            // pull tile it+2 toward L2/WGP$ (global_prefetch_b8)
            const int e0 = tid * 4;
            const long off = (long)((it + 2) * 64 + (e0 >> 6)) * D_ + (e0 & 63);
            __builtin_prefetch(kb + off, 0, 1);
            __builtin_prefetch(vb + off, 0, 1);
        }
        if (it + 1 < NT)
            stage((it + 1) & 1, nb + 64);               // overlap with compute below

        const bf16* Kt = sK [it & 1];
        const bf16* Vt = sVt[it & 1];

        // ---- S = Q * K^T  (16x64 strip per wave, 4 tiles x 2 K-chunks) ----
        f32x8 S[4];
#pragma unroll
        for (int t = 0; t < 4; ++t) S[t] = (f32x8){0,0,0,0,0,0,0,0};
#pragma unroll
        for (int t = 0; t < 4; ++t) {
#pragma unroll
            for (int dk = 0; dk < 2; ++dk) {
                const bf16* bp = &Kt[(t * 16 + l15) * 64 + dk * 32 + hi * 8];
                bf16x16 bK = cat16(lds8(bp), lds8(bp + 16));
                S[t] = __builtin_amdgcn_wmma_f32_16x16x32_bf16(
                           false, aQ[dk], false, bK, (short)0, S[t], false, false);
            }
        }

        // ---- P = relu(a*x^2 + b*x + g); row-sum; stage P (bf16) for GEMM2 ----
        bf16* Pw = sP[w];
#pragma unroll
        for (int t = 0; t < 4; ++t) {
#pragma unroll
            for (int r = 0; r < 8; ++r) {
                float x = S[t][r] * SCALE_;
                float p = fmaf(x, fmaf(al, x, be), ga);
                p = fmaxf(p, 0.0f);
                rs[r] += p;
                // C-layout element (row = r + hi*8, col = t*16 + l15) -> [m][n]
                Pw[(r + hi * 8) * 64 + t * 16 + l15] = (__bf16)p;
            }
        }
        asm volatile("s_wait_dscnt 0" ::: "memory");    // same-wave LDS RAW fence

        // ---- A-fragments of P (16x32 per K=32 chunk of the n dimension) ----
        bf16x16 aP[2];
#pragma unroll
        for (int nc = 0; nc < 2; ++nc) {
            const bf16* pp = &Pw[l15 * 64 + nc * 32 + c0];
            aP[nc] = cat16(lds8(pp), lds8(pp + 16));
        }

        // ---- O += P * V  (4 d-tiles x 2 n-chunks) ----
#pragma unroll
        for (int dt = 0; dt < 4; ++dt) {
#pragma unroll
            for (int nc = 0; nc < 2; ++nc) {
                const bf16* vp = &Vt[(dt * 16 + l15) * 64 + nc * 32 + hi * 8];
                bf16x16 bV = cat16(lds8(vp), lds8(vp + 16));
                O[dt] = __builtin_amdgcn_wmma_f32_16x16x32_bf16(
                            false, aP[nc], false, bV, (short)0, O[dt], false, false);
            }
        }
    }

    // ---- row-sum reduction across the 16 lanes sharing each row ----
#pragma unroll
    for (int r = 0; r < 8; ++r) {
        float s = rs[r];
        s += __shfl_xor(s, 1, 32);
        s += __shfl_xor(s, 2, 32);
        s += __shfl_xor(s, 4, 32);
        s += __shfl_xor(s, 8, 32);
        rs[r] = 1.0f / (s + EPS_);
    }

    // ---- normalize + store f32 ----
#pragma unroll
    for (int dt = 0; dt < 4; ++dt) {
#pragma unroll
        for (int r = 0; r < 8; ++r) {
            const int m = mt * 64 + w * 16 + r + hi * 8;
            ob[(long)m * D_ + dt * 16 + l15] = O[dt][r] * rs[r];
        }
    }
}

extern "C" void kernel_launch(void* const* d_in, const int* in_sizes, int n_in,
                              void* d_out, int out_size, void* d_ws, size_t ws_size,
                              hipStream_t stream) {
    const float* q     = (const float*)d_in[0];
    const float* k     = (const float*)d_in[1];
    const float* v     = (const float*)d_in[2];
    const float* alpha = (const float*)d_in[3];
    const float* beta  = (const float*)d_in[4];
    const float* gamma = (const float*)d_in[5];
    float* out = (float*)d_out;
    (void)in_sizes; (void)n_in; (void)out_size; (void)d_ws; (void)ws_size;

    dim3 grid(N_ / 64, H_, B_);
    dim3 block(128);
    l2q_attn_kernel<<<grid, block, 0, stream>>>(q, k, v, alpha, beta, gamma, out);
}
